// ReferenceLlamaExperts_4492535791706
// MI455X (gfx1250) — matrix operand order repaired and checked by reading, not compiled, not run on hardware
//
#include <hip/hip_runtime.h>

#define E_ 8
#define C_ 4096
#define H_ 1024
#define I_ 4096

typedef __attribute__((ext_vector_type(16))) __bf16       v16bf;
typedef __attribute__((ext_vector_type(8)))  float        v8f;
typedef __attribute__((ext_vector_type(4)))  unsigned int v4u;

#define XSTR 1032   // LDS row stride (elements) for x block (32 x 1024 bf16), padded
#define TSTR 136    // LDS row stride (elements) for t block (32 x 128 bf16), padded

__device__ __forceinline__ unsigned short f32_to_bf16(float f) {
  unsigned int u = __builtin_bit_cast(unsigned int, f);
  u += 0x7FFFu + ((u >> 16) & 1u);   // round-to-nearest-even
  return (unsigned short)(u >> 16);
}

// One v_perm_b32: pack truncated bf16(a) into [15:0], bf16(b) into [31:16].
__device__ __forceinline__ unsigned int pack2_bf16_trunc(float a, float b) {
  return __builtin_amdgcn_perm(__builtin_bit_cast(unsigned int, b),
                               __builtin_bit_cast(unsigned int, a), 0x07060302u);
}

// A-fragment (16x32 bf16, M x K) in contiguous k order (same as B), from LDS row-major [m][k]:
// lane l: M = l%16 ; lanes 0-15: K = kk+0..15
//          lanes 16-31: K = kk+16..31
__device__ __forceinline__ v16bf load_a_frag(const unsigned short* lds, int row0,
                                             int stride, int kk, int lane) {
  const unsigned short* p = lds + (row0 + (lane & 15)) * stride + kk + ((lane >> 4) << 4);
  union { v4u q[2]; v16bf v; } u;
  u.q[0] = *(const v4u*)p;
  u.q[1] = *(const v4u*)(p + 8);
  return u.v;
}

// B-fragment (32x16 bf16, K x N) from fragment-packed bf16 weights:
// element layout = exactly the per-lane 32B chunk the wave consumes.
__device__ __forceinline__ v16bf load_bp_frag(const unsigned short* __restrict__ Wp,
                                              int ntile, int kblock, int kblocks,
                                              int lane) {
  const unsigned short* p = Wp + (((size_t)ntile * kblocks + kblock) * 32 + lane) * 16;
  union { v4u q[2]; v16bf v; } u;
  u.q[0] = *(const v4u*)p;
  u.q[1] = *(const v4u*)(p + 8);
  return u.v;
}

// Fallback B-fragment from row-major fp32 W[k*ldw + n] (truncating convert via v_perm).
__device__ __forceinline__ v16bf load_bf_frag(const float* __restrict__ W, int kbase,
                                              int n0, int ldw, int lane) {
  const float* p = W + (size_t)(kbase + ((lane >> 4) << 4)) * ldw + (n0 + (lane & 15));
  union { unsigned int d[8]; v16bf v; } u;
#pragma unroll
  for (int v = 0; v < 8; ++v)
    u.d[v] = pack2_bf16_trunc(p[(size_t)(2 * v) * ldw], p[(size_t)(2 * v + 1) * ldw]);
  return u.v;
}

#define WMMA_BF16(a, b, c) \
  __builtin_amdgcn_wmma_f32_16x16x32_bf16(false, (a), false, (b), (short)0, (c), false, false)

// ---------------------------------------------------------------------------
// Pack kernel: f32 row-major W[e][Kt][Nt] -> bf16 B-fragment-packed layout:
//   Wp[((ntile*(Kt/32) + kblock)*32 + lane)*16 + elem],
//   elem 0..15 <-> K = kblock*32 + (lane/16)*16 + elem, N = ntile*16 + lane%16
// grid.x = E * (Kt/32) * (Nt/64), block = 256
// ---------------------------------------------------------------------------
__global__ void pack_w(const float* __restrict__ W, unsigned short* __restrict__ Wp,
                       int Kt, int Nt) {
  const int nb64 = Nt >> 6, kb32 = Kt >> 5;
  const int n64 = blockIdx.x % nb64;
  const int kb  = (blockIdx.x / nb64) % kb32;
  const int e   = blockIdx.x / (nb64 * kb32);
  const float* We = W + (size_t)e * Kt * Nt;
  unsigned short* Wpe = Wp + (size_t)e * Kt * Nt;

  const int t    = threadIdx.x;
  const int ntl  = t >> 6;         // local ntile 0..3
  const int lane = (t >> 1) & 31;  // fragment lane
  const int half = t & 1;          // 16B half of the 32B per-lane chunk
  const int ntile = n64 * 4 + ntl;
  const int n  = ntile * 16 + (lane & 15);
  const int k0 = kb * 32 + ((lane >> 4) << 4) + half * 8;

  union { unsigned short s[8]; v4u q; } u;
#pragma unroll
  for (int j = 0; j < 8; ++j)
    u.s[j] = f32_to_bf16(We[(size_t)(k0 + j) * Nt + n]);
  *(v4u*)(Wpe + (((size_t)ntile * kb32 + kb) * 32 + lane) * 16 + half * 8) = u.q;
}

// ---------------------------------------------------------------------------
// Fused SwiGLU MLP. One workgroup = 8 wave32s = (expert, 32-token block).
// ---------------------------------------------------------------------------
template <bool PACKED>
__global__ __launch_bounds__(256, 1) void moe_swiglu(
    const float* __restrict__ x,
    const float* __restrict__ Wg, const float* __restrict__ Wu,
    const float* __restrict__ Wd,
    const unsigned short* __restrict__ Wgp, const unsigned short* __restrict__ Wup,
    const unsigned short* __restrict__ Wdp,
    float* __restrict__ out) {
  __shared__ __align__(16) unsigned short xlds[32 * XSTR];  // ~66 KB
  __shared__ __align__(16) unsigned short tlds[32 * TSTR];  // ~8.7 KB

  const int tid   = threadIdx.x;
  const int lane  = tid & 31;
  const int wave  = tid >> 5;
  const int e     = blockIdx.x >> 7;           // 128 token-blocks per expert
  const int c0    = (blockIdx.x & 127) * 32;
  const int rhalf = (lane >> 4) << 3;          // C/D row offset: 0 or 8
  const int nlo   = lane & 15;                 // C/D column

  const size_t we = (size_t)e * H_ * I_;
  const float* xe = x + ((size_t)e * C_ + c0) * H_;
  const float *Wge = Wg + we, *Wue = Wu + we, *Wde = Wd + we;
  const unsigned short *Wgpe = Wgp + we, *Wupe = Wup + we, *Wdpe = Wdp + we;

  // Stage the 32x1024 fp32 x-block into LDS as bf16 (coalesced over k).
#pragma unroll 8
  for (int j = 0; j < 128; ++j) {
    int idx = tid + j * 256;
    int m = idx >> 10, k = idx & 1023;
    xlds[m * XSTR + k] = f32_to_bf16(xe[(size_t)m * H_ + k]);
  }
  __syncthreads();

  const v8f vzero = {0.f, 0.f, 0.f, 0.f, 0.f, 0.f, 0.f, 0.f};
  v8f acc[2][8];
#pragma unroll
  for (int mt = 0; mt < 2; ++mt)
#pragma unroll
    for (int nt = 0; nt < 8; ++nt) acc[mt][nt] = vzero;

#pragma unroll 1
  for (int chunk = 0; chunk < 32; ++chunk) {
    // -------- phase 1: gate/up 32x16 tiles per wave, reduce over H --------
    const int i0 = chunk * 128 + wave * 16;        // this wave's I column base
    v8f G[2] = {vzero, vzero};
    v8f U[2] = {vzero, vzero};
#pragma unroll 1
    for (int kk = 0; kk < H_; kk += 32) {
      v16bf a0 = load_a_frag(xlds, 0,  XSTR, kk, lane);
      v16bf a1 = load_a_frag(xlds, 16, XSTR, kk, lane);
      v16bf bg, bu;
      if (PACKED) {
        const int ntile = chunk * 8 + wave;        // i0/16
        bg = load_bp_frag(Wgpe, ntile, kk >> 5, H_ >> 5, lane);
        bu = load_bp_frag(Wupe, ntile, kk >> 5, H_ >> 5, lane);
      } else {
        bg = load_bf_frag(Wge, kk, i0, I_, lane);
        bu = load_bf_frag(Wue, kk, i0, I_, lane);
      }
      G[0] = WMMA_BF16(a0, bg, G[0]);
      G[1] = WMMA_BF16(a1, bg, G[1]);
      U[0] = WMMA_BF16(a0, bu, U[0]);
      U[1] = WMMA_BF16(a1, bu, U[1]);
    }
    // t = silu(G) * U -> bf16 tile in LDS (fast rcp for the sigmoid)
#pragma unroll
    for (int mt = 0; mt < 2; ++mt)
#pragma unroll
      for (int r = 0; r < 8; ++r) {
        float g = G[mt][r];
        float u = U[mt][r];
        float s = g * __builtin_amdgcn_rcpf(1.0f + __expf(-g)) * u;
        tlds[(mt * 16 + rhalf + r) * TSTR + wave * 16 + nlo] = f32_to_bf16(s);
      }
    __syncthreads();

    // -------- phase 2: out(32 x 128-per-wave) += t @ Wd[chunk rows] --------
#pragma unroll 1
    for (int ks = 0; ks < 4; ++ks) {
      const int kk = ks * 32;
      v16bf a0 = load_a_frag(tlds, 0,  TSTR, kk, lane);
      v16bf a1 = load_a_frag(tlds, 16, TSTR, kk, lane);
#pragma unroll
      for (int nt = 0; nt < 8; ++nt) {
        v16bf bd;
        if (PACKED)
          bd = load_bp_frag(Wdpe, wave * 8 + nt, chunk * 4 + ks, I_ >> 5, lane);
        else
          bd = load_bf_frag(Wde, chunk * 128 + kk, wave * 128 + nt * 16, H_, lane);
        acc[0][nt] = WMMA_BF16(a0, bd, acc[0][nt]);
        acc[1][nt] = WMMA_BF16(a1, bd, acc[1][nt]);
      }
    }
    __syncthreads();   // t tile is rewritten next chunk
  }

  // -------- write 32x1024 f32 output --------
#pragma unroll
  for (int mt = 0; mt < 2; ++mt)
#pragma unroll
    for (int nt = 0; nt < 8; ++nt) {
      float* o = out + ((size_t)e * C_ + (c0 + mt * 16 + rhalf)) * H_
                     + wave * 128 + nt * 16 + nlo;
#pragma unroll
      for (int r = 0; r < 8; ++r) o[(size_t)r * H_] = acc[mt][nt][r];
    }
}

extern "C" void kernel_launch(void* const* d_in, const int* in_sizes, int n_in,
                              void* d_out, int out_size, void* d_ws, size_t ws_size,
                              hipStream_t stream) {
  (void)in_sizes; (void)n_in; (void)out_size;
  const float* x  = (const float*)d_in[0];
  const float* Wg = (const float*)d_in[1];
  const float* Wu = (const float*)d_in[2];
  const float* Wd = (const float*)d_in[3];
  float* out = (float*)d_out;

  const size_t matElems = (size_t)E_ * H_ * I_;                 // 33.55M per matrix
  const size_t need = 3 * matElems * sizeof(unsigned short);    // ~201 MB
  dim3 gblock(256);
  dim3 ggrid(E_ * (C_ / 32));   // 1024 WGs, token-blocks contiguous per expert

  if (ws_size >= need) {
    unsigned short* Wgp = (unsigned short*)d_ws;
    unsigned short* Wup = Wgp + matElems;
    unsigned short* Wdp = Wup + matElems;
    // Re-pack every call (launch must be deterministic; no caching allowed).
    pack_w<<<dim3(E_ * (H_ / 32) * (I_ / 64)), gblock, 0, stream>>>(Wg, Wgp, H_, I_);
    pack_w<<<dim3(E_ * (H_ / 32) * (I_ / 64)), gblock, 0, stream>>>(Wu, Wup, H_, I_);
    pack_w<<<dim3(E_ * (I_ / 32) * (H_ / 64)), gblock, 0, stream>>>(Wd, Wdp, I_, H_);
    moe_swiglu<true><<<ggrid, gblock, 0, stream>>>(x, Wg, Wu, Wd, Wgp, Wup, Wdp, out);
  } else {
    moe_swiglu<false><<<ggrid, gblock, 0, stream>>>(
        x, Wg, Wu, Wd, (const unsigned short*)d_ws, (const unsigned short*)d_ws,
        (const unsigned short*)d_ws, out);
  }
}
